// MultiHeadAttention_6287832121433
// MI455X (gfx1250) — matrix operand order repaired and checked by reading, not compiled, not run on hardware
//
#include <hip/hip_runtime.h>
#include <hip/hip_bf16.h>

typedef __attribute__((ext_vector_type(16))) _Float16 v16h;
typedef __attribute__((ext_vector_type(8)))  _Float16 v8h;
typedef __attribute__((ext_vector_type(8)))  float    v8f;

#define DEV __device__ __forceinline__

constexpr int B  = 2;
constexpr int S  = 2048;
constexpr int D  = 1024;
constexpr int H  = 16;
constexpr int DK = 64;
constexpr int M  = B * S;   // 4096 flattened rows

// ---------------------------------------------------------------------------
// CDNA5 WMMA fragment helpers (wave32). Layouts per CDNA5 ISA.
// Every loader is two contiguous 16-byte loads per lane.
// ---------------------------------------------------------------------------

// A fragment 16x32 f16 from row-major src (leading dim lda).
DEV v16h load_frag_a(const _Float16* src, int lda) {
  const int lane = threadIdx.x & 31;
  const _Float16* p = src + (lane & 15) * lda + ((lane >> 4) << 3);
  v8h lo = *(const v8h*)p;          // K base+0..7   (pairs)
  v8h hi = *(const v8h*)(p + 16);   // K base+16..23
  return __builtin_shufflevector(lo, hi, 0, 1, 2, 3, 4, 5, 6, 7,
                                 8, 9, 10, 11, 12, 13, 14, 15);
}

// B fragment 32x16 (KxN) where logical B[k][n] = src[n*ld + k]
DEV v16h load_frag_bt(const _Float16* src, int ld) {
  const int lane = threadIdx.x & 31;
  const _Float16* p = src + (lane & 15) * ld + ((lane >> 4) << 3);
  v8h lo = *(const v8h*)p;
  v8h hi = *(const v8h*)(p + 16);
  return __builtin_shufflevector(lo, hi, 0, 1, 2, 3, 4, 5, 6, 7,
                                 8, 9, 10, 11, 12, 13, 14, 15);
}

DEV v8f wmma_f16(v16h a, v16h b, v8f c) {
  return __builtin_amdgcn_wmma_f32_16x16x32_f16(false, a, false, b, (short)0, c,
                                                false, false);
}

DEV float wave_max(float v) {
#pragma unroll
  for (int off = 16; off >= 1; off >>= 1) v = fmaxf(v, __shfl_xor(v, off, 32));
  return v;
}
DEV float wave_sum(float v) {
#pragma unroll
  for (int off = 16; off >= 1; off >>= 1) v += __shfl_xor(v, off, 32);
  return v;
}

// ---------------------------------------------------------------------------
// Kernel 1a: fp32 -> f16 convert (activations)
// ---------------------------------------------------------------------------
__global__ void cvt_kernel(const float* __restrict__ in, _Float16* __restrict__ out,
                           int n) {
  int i = blockIdx.x * blockDim.x + threadIdx.x;
  int stride = gridDim.x * blockDim.x;
  for (; i < n; i += stride) out[i] = (_Float16)in[i];
}

// ---------------------------------------------------------------------------
// Kernel 1b: fp32 -> f16 convert + transpose (weights): out[n][k] = in[k][n]
// ---------------------------------------------------------------------------
__global__ void __launch_bounds__(256)
cvtT_kernel(const float* __restrict__ in, _Float16* __restrict__ out) {
  __shared__ _Float16 tile[32][33];
  const int k0 = blockIdx.x * 32, n0 = blockIdx.y * 32;
  const int tx = threadIdx.x & 31, ty = threadIdx.x >> 5;  // ty 0..7
#pragma unroll
  for (int r = ty; r < 32; r += 8)
    tile[r][tx] = (_Float16)in[(size_t)(k0 + r) * D + n0 + tx];
  __syncthreads();
#pragma unroll
  for (int r = ty; r < 32; r += 8)
    out[(size_t)(n0 + r) * D + k0 + tx] = tile[tx][r];
}

// ---------------------------------------------------------------------------
// Kernel 2: projection GEMM  out = X[M,D] @ W + bias, W transposed WT[N][K].
// vmode 0: out f16 [B,H,S,dk] (Q,K); vmode 1: out f16 [B,H,dk,S] (V^T).
// grid (M/64, D/128). Wave w: cols blockIdx.y*128 + w*16, 4 row-tiles ->
// 4 WMMAs/K-step. 2x-unrolled software pipeline with ping-pong register
// sets: loads of one set overlap WMMAs of the other, no rotation movs.
// ---------------------------------------------------------------------------
__global__ void __launch_bounds__(256)
proj_kernel(const _Float16* __restrict__ X, const _Float16* __restrict__ WT,
            const float* __restrict__ bias, _Float16* __restrict__ out,
            int vmode) {
  const int row0 = blockIdx.x * 64;
  const int nb   = blockIdx.y * 128;
  const int w    = threadIdx.x >> 5;
  const int lane = threadIdx.x & 31;

  const _Float16* Wcol = WT + (size_t)(nb + w * 16) * D;  // 16 cols, K-contiguous
  const _Float16* Arow = X + (size_t)row0 * D;

  v8f acc[4] = {};
  v16h b0 = load_frag_bt(Wcol, D);
  v16h a0[4];
#pragma unroll
  for (int rt = 0; rt < 4; ++rt)
    a0[rt] = load_frag_a(Arow + (size_t)(rt * 16) * D, D);

  for (int i = 0; i < D / 64 - 1; ++i) {  // 15 iterations, 2 K-steps each
    const int kA = i * 64 + 32;
    const int kB = i * 64 + 64;
    v16h b1 = load_frag_bt(Wcol + kA, D);
    v16h a1[4];
#pragma unroll
    for (int rt = 0; rt < 4; ++rt)
      a1[rt] = load_frag_a(Arow + (size_t)(rt * 16) * D + kA, D);
#pragma unroll
    for (int rt = 0; rt < 4; ++rt) acc[rt] = wmma_f16(a0[rt], b0, acc[rt]);

    b0 = load_frag_bt(Wcol + kB, D);
#pragma unroll
    for (int rt = 0; rt < 4; ++rt)
      a0[rt] = load_frag_a(Arow + (size_t)(rt * 16) * D + kB, D);
#pragma unroll
    for (int rt = 0; rt < 4; ++rt) acc[rt] = wmma_f16(a1[rt], b1, acc[rt]);
  }
  {  // epilogue: steps D-64 (in set0) and D-32
    const int kA = D - 32;
    v16h b1 = load_frag_bt(Wcol + kA, D);
    v16h a1[4];
#pragma unroll
    for (int rt = 0; rt < 4; ++rt)
      a1[rt] = load_frag_a(Arow + (size_t)(rt * 16) * D + kA, D);
#pragma unroll
    for (int rt = 0; rt < 4; ++rt) acc[rt] = wmma_f16(a0[rt], b0, acc[rt]);
#pragma unroll
    for (int rt = 0; rt < 4; ++rt) acc[rt] = wmma_f16(a1[rt], b1, acc[rt]);
  }

  const int nloc = lane & 15;
  const int hi   = lane >> 4;
  const int n    = nb + w * 16 + nloc;
  const float bval = bias[n];
  const int h = n >> 6, dcol = n & 63;
#pragma unroll
  for (int rt = 0; rt < 4; ++rt) {
#pragma unroll
    for (int v = 0; v < 8; ++v) {
      int row = row0 + rt * 16 + v + 8 * hi;
      int bb  = row >> 11;        // / S
      int s   = row & (S - 1);
      float r = acc[rt][v] + bval;
      if (vmode == 0)
        out[(((size_t)bb * H + h) * S + s) * DK + dcol] = (_Float16)r;
      else
        out[(((size_t)bb * H + h) * DK + dcol) * S + s] = (_Float16)r;
    }
  }
}

// ---------------------------------------------------------------------------
// Kernel 3: scores = Q @ K^T / sqrt(dk) -> attn (fp32, [B,H,S,S])
// grid (S/16, S/256, B*H); block shares one Q tile via LDS,
// each wave computes two 16x16 score tiles (4 WMMAs).
// ---------------------------------------------------------------------------
__global__ void __launch_bounds__(256)
scores_kernel(const _Float16* __restrict__ Q, const _Float16* __restrict__ K,
              float* __restrict__ attn) {
  __shared__ alignas(16) _Float16 Qs[16 * DK];  // 2KB
  const int s0 = blockIdx.x * 16;
  const int bh = blockIdx.z;
  const int w  = threadIdx.x >> 5;
  const int t0 = blockIdx.y * 256 + w * 32;
  const int lane = threadIdx.x & 31;

  const _Float16* Qg = Q + ((size_t)bh * S + s0) * DK;
  ((unsigned long long*)Qs)[threadIdx.x] =
      ((const unsigned long long*)Qg)[threadIdx.x];  // 256 * 8B = 2KB
  __syncthreads();

  const _Float16* Kt = K + ((size_t)bh * S + t0) * DK;  // rows t, contiguous d
  v16h a0 = load_frag_a(Qs, DK);
  v16h a1 = load_frag_a(Qs + 32, DK);

  v8f c0 = {}, c1 = {};
  c0 = wmma_f16(a0, load_frag_bt(Kt, DK), c0);
  c0 = wmma_f16(a1, load_frag_bt(Kt + 32, DK), c0);
  const _Float16* Kt2 = Kt + (size_t)16 * DK;
  c1 = wmma_f16(a0, load_frag_bt(Kt2, DK), c1);
  c1 = wmma_f16(a1, load_frag_bt(Kt2 + 32, DK), c1);

  const int nloc = lane & 15, hi = lane >> 4;
  float* orow = attn + ((size_t)bh * S + s0) * S + t0;
#pragma unroll
  for (int v = 0; v < 8; ++v) {
    int m = v + 8 * hi;
    orow[(size_t)m * S + nloc]      = c0[v] * 0.125f;  // 1/sqrt(64)
    orow[(size_t)m * S + 16 + nloc] = c1[v] * 0.125f;
  }
}

// ---------------------------------------------------------------------------
// Kernel 4: in-place row softmax over attn, one wave per 2048-wide row.
// ---------------------------------------------------------------------------
__global__ void __launch_bounds__(256)
softmax_kernel(float* __restrict__ attn) {
  __shared__ float buf[8][S];  // 64KB
  const int w = threadIdx.x >> 5, lane = threadIdx.x & 31;
  float* p = attn + ((size_t)blockIdx.x * 8 + w) * S;

  float mx = -1e30f;
  for (int i = lane; i < S; i += 32) {
    float v = p[i];
    buf[w][i] = v;
    mx = fmaxf(mx, v);
  }
  mx = wave_max(mx);
  float sum = 0.f;
  for (int i = lane; i < S; i += 32) {
    float e = __expf(buf[w][i] - mx);
    buf[w][i] = e;
    sum += e;
  }
  sum = wave_sum(sum);
  float inv = 1.0f / sum;
  for (int i = lane; i < S; i += 32) p[i] = buf[w][i] * inv;
}

// ---------------------------------------------------------------------------
// Kernel 5: ctx[B,S,D] (f16) = attn @ V, V transposed Vt[B,H,dk,S].
// grid (S/64, B*H); block tile 64 rows x 64 cols, 2 WMMAs/wave/K-step.
// Double-buffered LDS staging (fp32->f16): one barrier per K-step,
// next tile staged while current tile feeds the WMMAs.
// ---------------------------------------------------------------------------
__global__ void __launch_bounds__(256)
ctx_kernel(const float* __restrict__ attn, const _Float16* __restrict__ Vt,
           _Float16* __restrict__ ctx) {
  __shared__ alignas(16) _Float16 As[2][64 * 32];  // 2 x 4KB
  const int bh = blockIdx.y;
  const int bb = bh >> 4, h = bh & 15;
  const int row0 = blockIdx.x * 64;
  const int w = threadIdx.x >> 5;
  const int lane = threadIdx.x & 31;
  const int rt0 = (w >> 2) * 2;  // first of two row tiles
  const int ct = w & 3;

  const float* Abase = attn + ((size_t)bh * S + row0) * S;
  const _Float16* Vcol = Vt + ((size_t)bh * DK + ct * 16) * S;  // 16 d-cols, t-contig

  auto stage = [&](int bufi, int k0) {
    int idx = threadIdx.x * 8;
    int r = idx >> 5, kk = idx & 31;
    const float* g = &Abase[(size_t)r * S + k0 + kk];
    float4 x = *(const float4*)g;
    float4 yv = *(const float4*)(g + 4);
    v8h hh;
    hh[0] = (_Float16)x.x;  hh[1] = (_Float16)x.y;
    hh[2] = (_Float16)x.z;  hh[3] = (_Float16)x.w;
    hh[4] = (_Float16)yv.x; hh[5] = (_Float16)yv.y;
    hh[6] = (_Float16)yv.z; hh[7] = (_Float16)yv.w;
    *(v8h*)&As[bufi][idx] = hh;
  };

  stage(0, 0);
  __syncthreads();

  v8f acc0 = {}, acc1 = {};
  int cur = 0;
  for (int k0 = 0; k0 < S; k0 += 32) {
    if (k0 + 32 < S) stage(cur ^ 1, k0 + 32);  // overlaps with compute below
    v16h b = load_frag_bt(Vcol + k0, S);
    v16h a0 = load_frag_a(&As[cur][(rt0 * 16) * 32], 32);
    v16h a1 = load_frag_a(&As[cur][((rt0 + 1) * 16) * 32], 32);
    acc0 = wmma_f16(a0, b, acc0);
    acc1 = wmma_f16(a1, b, acc1);
    __syncthreads();
    cur ^= 1;
  }

  const int nloc = lane & 15, hi = lane >> 4;
#pragma unroll
  for (int v = 0; v < 8; ++v) {
    int s0 = row0 + rt0 * 16 + v + 8 * hi;
    int s1 = s0 + 16;
    ctx[((size_t)bb * S + s0) * D + h * DK + ct * 16 + nloc] = (_Float16)acc0[v];
    ctx[((size_t)bb * S + s1) * D + h * DK + ct * 16 + nloc] = (_Float16)acc1[v];
  }
}

// ---------------------------------------------------------------------------
// Kernel 6: y = LayerNorm(ctx @ Wo + bo + query) * gamma + beta  (WoT transposed)
// grid (M/16). Wave w owns cols [w*128, w*128+128): 8 acc tiles. All 8 B
// fragments of a K-step loaded as one clause; next A fragment prefetched.
// ---------------------------------------------------------------------------
__global__ void __launch_bounds__(256)
outln_kernel(const _Float16* __restrict__ ctx, const _Float16* __restrict__ WoT,
             const float* __restrict__ bo, const float* __restrict__ query,
             const float* __restrict__ gamma, const float* __restrict__ beta,
             float* __restrict__ y) {
  __shared__ float smem[16 * D];  // 64KB row buffer
  const int row0 = blockIdx.x * 16;
  const int w = threadIdx.x >> 5;
  const int lane = threadIdx.x & 31;

  const _Float16* Wbase = WoT + (size_t)(w * 128) * D;
  const _Float16* Arow  = ctx + (size_t)row0 * D;

  v8f acc[8] = {};
  v16h a = load_frag_a(Arow, D);
  for (int k0 = 0; k0 < D; k0 += 32) {
    v16h bf[8];
#pragma unroll
    for (int jt = 0; jt < 8; ++jt)
      bf[jt] = load_frag_bt(Wbase + (size_t)(jt * 16) * D + k0, D);
    v16h acur = a;
    if (k0 + 32 < D) a = load_frag_a(Arow + k0 + 32, D);  // prefetch next A
#pragma unroll
    for (int jt = 0; jt < 8; ++jt) acc[jt] = wmma_f16(acur, bf[jt], acc[jt]);
  }

  const int nloc = lane & 15, hi = lane >> 4;
#pragma unroll
  for (int jt = 0; jt < 8; ++jt) {
    int n = w * 128 + jt * 16 + nloc;
    float bval = bo[n];
#pragma unroll
    for (int v = 0; v < 8; ++v) {
      int m = v + 8 * hi;
      smem[m * D + n] = acc[jt][v] + bval + query[(size_t)(row0 + m) * D + n];
    }
  }
  __syncthreads();

#pragma unroll
  for (int rr = 0; rr < 2; ++rr) {
    int r = w * 2 + rr;
    const float* rb = smem + r * D;
    float s1 = 0.f, s2 = 0.f;
    for (int i = lane; i < D; i += 32) {
      float x = rb[i];
      s1 += x;
      s2 += x * x;
    }
    s1 = wave_sum(s1);
    s2 = wave_sum(s2);
    float mu  = s1 * (1.0f / D);
    float var = s2 * (1.0f / D) - mu * mu;
    float rstd = rsqrtf(var + 1e-5f);
    float* yr = y + (size_t)(row0 + r) * D;
    for (int i = lane; i < D; i += 32)
      yr[i] = (rb[i] - mu) * rstd * gamma[i] + beta[i];
  }
}

// ---------------------------------------------------------------------------
// Host launcher
// ---------------------------------------------------------------------------
extern "C" void kernel_launch(void* const* d_in, const int* in_sizes, int n_in,
                              void* d_out, int out_size, void* d_ws, size_t ws_size,
                              hipStream_t stream) {
  const float* query = (const float*)d_in[0];
  const float* key   = (const float*)d_in[1];
  const float* value = (const float*)d_in[2];
  const float* Wq = (const float*)d_in[3];
  const float* bq = (const float*)d_in[4];
  const float* Wk = (const float*)d_in[5];
  const float* bk = (const float*)d_in[6];
  const float* Wv = (const float*)d_in[7];
  const float* bv = (const float*)d_in[8];
  const float* Wo = (const float*)d_in[9];
  const float* bo = (const float*)d_in[10];
  const float* gamma = (const float*)d_in[11];
  const float* beta  = (const float*)d_in[12];

  float* y    = (float*)d_out;
  float* attn = y + (size_t)M * D;  // [B,H,S,S] fp32, second tuple output

  const size_t NXD = (size_t)M * D;   // 4M
  const size_t NW  = (size_t)D * D;   // 1M
  _Float16* ws = (_Float16*)d_ws;
  _Float16* xq = ws;
  _Float16* xk = xq + NXD;
  _Float16* xv = xk + NXD;
  _Float16* wqT = xv + NXD;
  _Float16* wkT = wqT + NW;
  _Float16* wvT = wkT + NW;
  _Float16* woT = wvT + NW;
  _Float16* Qh  = woT + NW;
  _Float16* Kh  = Qh + NXD;
  _Float16* VhT = Kh + NXD;
  _Float16* Ch  = VhT + NXD;

  cvt_kernel<<<2048, 256, 0, stream>>>(query, xq, (int)NXD);
  cvt_kernel<<<2048, 256, 0, stream>>>(key,   xk, (int)NXD);
  cvt_kernel<<<2048, 256, 0, stream>>>(value, xv, (int)NXD);

  dim3 gT(D / 32, D / 32);
  cvtT_kernel<<<gT, 256, 0, stream>>>(Wq, wqT);
  cvtT_kernel<<<gT, 256, 0, stream>>>(Wk, wkT);
  cvtT_kernel<<<gT, 256, 0, stream>>>(Wv, wvT);
  cvtT_kernel<<<gT, 256, 0, stream>>>(Wo, woT);

  dim3 gProj(M / 64, D / 128);
  proj_kernel<<<gProj, 256, 0, stream>>>(xq, wqT, bq, Qh, 0);
  proj_kernel<<<gProj, 256, 0, stream>>>(xk, wkT, bk, Kh, 0);
  proj_kernel<<<gProj, 256, 0, stream>>>(xv, wvT, bv, VhT, 1);

  dim3 gSc(S / 16, S / 256, B * H);
  scores_kernel<<<gSc, 256, 0, stream>>>(Qh, Kh, attn);

  softmax_kernel<<<(B * H * S) / 8, 256, 0, stream>>>(attn);

  dim3 gCtx(S / 64, B * H);
  ctx_kernel<<<gCtx, 256, 0, stream>>>(attn, VhT, Ch);

  outln_kernel<<<M / 16, 256, 0, stream>>>(Ch, woT, bo, query, gamma, beta, y);
}
